// SAN_decoder_49898930045480
// MI455X (gfx1250) — matrix operand order repaired and checked by reading, not compiled, not run on hardware
//
#include <hip/hip_runtime.h>
#include <math.h>

typedef _Float16 v16h __attribute__((ext_vector_type(16)));
typedef float    v8f  __attribute__((ext_vector_type(8)));

#define C_DIM   684
#define H_DIM   256
#define A_DIM   512
#define V_DIM   415
#define CCOV    256
#define KTAP    11
#define HFD     32
#define WFD     128
#define NPOS    4096
#define TSTEPS  64
#define KPAD    128   // 121 taps padded to 128 (4 x K=32 WMMA steps)

// hardware tanh (V_TANH_F32, gfx1250 trans op) with safe fallback
#if __has_builtin(__builtin_amdgcn_tanhf)
__device__ __forceinline__ float fast_tanh(float x) { return __builtin_amdgcn_tanhf(x); }
#else
__device__ __forceinline__ float fast_tanh(float x) { return tanhf(x); }
#endif

// ---------------- workspace layout (float offsets) ----------------
static constexpr size_t OFF_CNNT   = 0;                         // 4096*512
static constexpr size_t OFF_KEFF   = OFF_CNNT   + 4096ull*512;  // 128*512 f32
static constexpr size_t OFF_KFRAG  = OFF_KEFF   + 128ull*512;   // 65536 halves = 32768 f
static constexpr size_t OFF_WIHTI  = OFF_KFRAG  + 32768;        // 256*768
static constexpr size_t OFF_WHHTI  = OFF_WIHTI  + 196608;       // 256*768
static constexpr size_t OFF_WIHTO  = OFF_WHHTI  + 196608;       // 684*768
static constexpr size_t OFF_WHHTO  = OFF_WIHTO  + 525312;       // 256*768
static constexpr size_t OFF_WQT    = OFF_WHHTO  + 196608;       // 256*512
static constexpr size_t OFF_WST    = OFF_WQT    + 131072;       // 256*128
static constexpr size_t OFF_WET    = OFF_WST    + 32768;        // 256*128
static constexpr size_t OFF_WCT    = OFF_WET    + 32768;        // 684*128
static constexpr size_t OFF_WVT    = OFF_WCT    + 87552;        // 128*415
static constexpr size_t OFF_ASUM   = OFF_WVT    + 53120;        // 4096
static constexpr size_t OFF_ALPHA  = OFF_ASUM   + 4096;
static constexpr size_t OFF_ENERGY = OFF_ALPHA  + 4096;
static constexpr size_t OFF_MASKV  = OFF_ENERGY + 4096;
static constexpr size_t OFF_AVG    = OFF_MASKV  + 4096;         // 684
static constexpr size_t OFF_H      = OFF_AVG    + 704;          // 256
static constexpr size_t OFF_H1     = OFF_H      + 256;
static constexpr size_t OFF_EMB    = OFF_H1     + 256;
static constexpr size_t OFF_Q      = OFF_EMB    + 256;          // 512
static constexpr size_t OFF_CTX    = OFF_Q      + 512;          // 684

// ---------------- generic transpose: out[c*R+r] = in[r*C+c] ----------------
__global__ void k_transpose(const float* __restrict__ in, float* __restrict__ out,
                            int R, int Cc) {
    int gid = blockIdx.x * 256 + threadIdx.x;
    if (gid >= R * Cc) return;
    int r = gid / Cc, c = gid - r * Cc;
    out[c * R + r] = in[gid];
}

// mask subsample: maskv[pos] = images_mask[y*16, x*16]
__global__ void k_mask(const float* __restrict__ im, float* __restrict__ maskv) {
    int gid = blockIdx.x * 256 + threadIdx.x;
    if (gid >= NPOS) return;
    int y = gid >> 7, x = gid & 127;
    maskv[gid] = im[(y * 16) * (WFD * 16) + x * 16];
}

// masked average per channel
__global__ void k_avg(const float* __restrict__ feat, const float* __restrict__ maskv,
                      float* __restrict__ avg) {
    __shared__ float sN[256], sD[256];
    int c = blockIdx.x, t = threadIdx.x;
    float n = 0.f, d = 0.f;
    for (int p = t; p < NPOS; p += 256) {
        float m = maskv[p];
        n += feat[c * NPOS + p] * m;
        d += m;
    }
    sN[t] = n; sD[t] = d; __syncthreads();
    for (int s = 128; s > 0; s >>= 1) {
        if (t < s) { sN[t] += sN[t + s]; sD[t] += sD[t + s]; }
        __syncthreads();
    }
    if (t == 0) avg[c] = sN[0] / sD[0];
}

__global__ void k_h0(const float* __restrict__ avg, const float* __restrict__ Wi,
                     const float* __restrict__ bi, float* __restrict__ h) {
    __shared__ float sA[C_DIM];
    int t = threadIdx.x;
    for (int i = t; i < C_DIM; i += 256) sA[i] = avg[i];
    __syncthreads();
    float s = bi[t];
    for (int c = 0; c < C_DIM; ++c) s += sA[c] * Wi[t * C_DIM + c];
    h[t] = fast_tanh(s);
}

__global__ void k_init(const float* __restrict__ emb_table,
                       float* __restrict__ asum, float* __restrict__ emb) {
    int gid = blockIdx.x * 256 + threadIdx.x;
    if (gid < NPOS) asum[gid] = 0.f;
    if (gid < H_DIM) emb[gid] = emb_table[1 * H_DIM + gid];  // word id 1
}

// Keff[t][a] = sum_c Wcov[a][c] * conv_k[c][t]   (t padded to 128 with zeros)
__global__ void k_keff(const float* __restrict__ convk, const float* __restrict__ Wcov,
                       float* __restrict__ keff) {
    int gid = blockIdx.x * 256 + threadIdx.x;
    if (gid >= KPAD * A_DIM) return;
    int t = gid >> 9, a = gid & 511;
    float s = 0.f;
    if (t < KTAP * KTAP)
        for (int c = 0; c < CCOV; ++c)
            s += convk[c * (KTAP * KTAP) + t] * Wcov[a * CCOV + c];
    keff[gid] = s;
}

// swizzle Keff into per-lane WMMA B fragments (f16):
// flat = ((nt*4+kk)*32 + lane)*16 + i ; half i of lane => k = kk*32 + i + 16*(lane>>4), n = nt*16 + (lane&15)
__global__ void k_kfrag(const float* __restrict__ keff, _Float16* __restrict__ kfrag) {
    int gid = blockIdx.x * 256 + threadIdx.x;
    if (gid >= 32 * 4 * 32 * 16) return;
    int i = gid & 15, lane = (gid >> 4) & 31, kk = (gid >> 9) & 3, nt = gid >> 11;
    int k = kk * 32 + i + 16 * (lane >> 4);
    int n = nt * 16 + (lane & 15);
    kfrag[gid] = (_Float16)keff[k * A_DIM + n];
}

// cnn_trans[pos][a] = sum_c feat[c][pos]*Wproj[a][c] + bproj[a]
// grid (16, 32): 256 pos per block, 16 a per block
__global__ void k_cnntrans(const float* __restrict__ feat, const float* __restrict__ Wproj,
                           const float* __restrict__ bproj, float* __restrict__ ct) {
    __shared__ float sW[128 * 16];
    int tid = threadIdx.x;
    int pos = blockIdx.x * 256 + tid;
    int a0 = blockIdx.y * 16;
    float acc[16];
#pragma unroll
    for (int j = 0; j < 16; ++j) acc[j] = 0.f;
    for (int cc = 0; cc < C_DIM; cc += 128) {
        int chunk = (C_DIM - cc < 128) ? (C_DIM - cc) : 128;
        for (int idx = tid; idx < chunk * 16; idx += 256) {
            int j = idx >> 4, aa = idx & 15;
            sW[idx] = Wproj[(a0 + aa) * C_DIM + cc + j];
        }
        __syncthreads();
        for (int j = 0; j < chunk; ++j) {
            float f = feat[(cc + j) * NPOS + pos];
#pragma unroll
            for (int aa = 0; aa < 16; ++aa) acc[aa] += f * sW[(j << 4) + aa];
        }
        __syncthreads();
    }
#pragma unroll
    for (int aa = 0; aa < 16; ++aa)
        ct[pos * A_DIM + a0 + aa] = acc[aa] + bproj[a0 + aa];
}

__device__ __forceinline__ float sigm(float x) { return 1.f / (1.f + expf(-x)); }

// input GRU + q projection (single workgroup; weights pre-transposed for coalescing)
__global__ void k_gru_in(const float* __restrict__ WihT, const float* __restrict__ WhhT,
                         const float* __restrict__ bih, const float* __restrict__ bhh,
                         const float* __restrict__ WqT, const float* __restrict__ bq,
                         const float* __restrict__ emb, const float* __restrict__ hprev,
                         float* __restrict__ h1out, float* __restrict__ qout) {
    __shared__ float sE[H_DIM], sH[H_DIM], sH1[H_DIM];
    int t = threadIdx.x;
    sE[t] = emb[t]; sH[t] = hprev[t];
    __syncthreads();
    float gr = bih[t], gz = bih[t + 256], gn = bih[t + 512];
    float hr = bhh[t], hz = bhh[t + 256], hn = bhh[t + 512];
    for (int c = 0; c < H_DIM; ++c) {
        float e = sE[c], hv = sH[c];
        const float* wi = WihT + c * 768;
        const float* wh = WhhT + c * 768;
        gr += e * wi[t]; gz += e * wi[t + 256]; gn += e * wi[t + 512];
        hr += hv * wh[t]; hz += hv * wh[t + 256]; hn += hv * wh[t + 512];
    }
    float r = sigm(gr + hr), z = sigm(gz + hz);
    float n = fast_tanh(gn + r * hn);
    float h1 = (1.f - z) * n + z * sH[t];
    sH1[t] = h1; h1out[t] = h1;
    __syncthreads();
    for (int i = t; i < A_DIM; i += 256) {
        float s = bq[i];
        for (int c = 0; c < H_DIM; ++c) s += sH1[c] * WqT[c * A_DIM + i];
        qout[i] = s;
    }
}

// fused conv+proj attention energy: WMMA GEMM patch(16x128) x Keff(128x512)
// block = 16 positions (one row segment), 8 waves sweep N-tiles.
// C operand of the WMMA chain is pre-loaded with q[n] + cnn_trans[pos][n] so the
// epilogue is just tanh (hardware V_TANH_F32) * wa -> LDS atomic reduce.
__global__ void k_attend(const _Float16* __restrict__ kfrag, const float* __restrict__ asum,
                         const float* __restrict__ q, const float* __restrict__ ct,
                         const float* __restrict__ wa, const float* __restrict__ ba,
                         const float* __restrict__ maskv, float* __restrict__ energy) {
    __shared__ float sPatch[KTAP][32];
    __shared__ float sEnergy[16];
    int tid = threadIdx.x;
    int pos0 = blockIdx.x * 16;
    int y = pos0 >> 7, x0 = pos0 & 127;

    for (int idx = tid; idx < KTAP * 26; idx += 256) {
        int row = idx / 26, col = idx - row * 26;
        int gy = y + row - 5, gx = x0 + col - 5;
        sPatch[row][col] =
            (gy >= 0 && gy < HFD && gx >= 0 && gx < WFD) ? asum[gy * WFD + gx] : 0.f;
    }
    if (tid < 16) sEnergy[tid] = 0.f;
    __syncthreads();

    int lane = tid & 31, wv = tid >> 5;
    int hh = lane >> 4, m = lane & 15;

    // A fragments: 16x32 f16, k = tap index (im2col of alpha_sum), reused for all N tiles
    v16h af[4];
#pragma unroll
    for (int kk = 0; kk < 4; ++kk) {
#pragma unroll
        for (int i = 0; i < 16; ++i) {
            int v = i >> 1, j = i & 1;
            int k = kk * 32 + i + 16 * hh;
            _Float16 val = (_Float16)0.f;
            if (k < KTAP * KTAP) {
                int dy = k / KTAP, dx = k - dy * KTAP;
                val = (_Float16)sPatch[dy][m + dx];
            }
            af[kk][i] = val;
        }
    }

    for (int nt = wv; nt < 32; nt += 8) {
        if (nt + 8 < 32)
            __builtin_prefetch(kfrag + (((nt + 8) * 4) * 32 + lane) * 16, 0, 1);
        int n = (nt << 4) + m;
        float qn = q[n];
        // init C with bias terms: issues ct loads ahead of the WMMA chain
        v8f acc;
#pragma unroll
        for (int r = 0; r < 8; ++r)
            acc[r] = qn + ct[(pos0 + r + 8 * hh) * A_DIM + n];
#pragma unroll
        for (int kk = 0; kk < 4; ++kk) {
            v16h bf = *(const v16h*)(kfrag + (((nt * 4 + kk) * 32 + lane) << 4));
            acc = __builtin_amdgcn_wmma_f32_16x16x32_f16(
                false, af[kk], false, bf, (short)0, acc, false, false);
        }
        float wan = wa[n];
#pragma unroll
        for (int r = 0; r < 8; ++r) {
            int mr = r + 8 * hh;
            float e = fast_tanh(acc[r]);
            atomicAdd(&sEnergy[mr], wan * e);
        }
    }
    __syncthreads();
    if (tid < 16) {
        int pos = pos0 + tid;
        float en = sEnergy[tid] + ba[0];
        energy[pos] = (maskv[pos] > 0.f) ? en : -1.0e9f;
    }
}

// softmax over 4096 + alpha_sum accumulation (single block, 1024 threads)
__global__ void k_softmax(const float* __restrict__ energy, float* __restrict__ alpha,
                          float* __restrict__ asum) {
    __shared__ float red[1024];
    int t = threadIdx.x;
    float e[4];
    float mx = -3.0e38f;
#pragma unroll
    for (int j = 0; j < 4; ++j) {
        e[j] = energy[t + j * 1024];
        mx = fmaxf(mx, e[j]);
    }
    red[t] = mx; __syncthreads();
    for (int s = 512; s > 0; s >>= 1) {
        if (t < s) red[t] = fmaxf(red[t], red[t + s]);
        __syncthreads();
    }
    float gmax = red[0]; __syncthreads();
    float ex[4], ssum = 0.f;
#pragma unroll
    for (int j = 0; j < 4; ++j) { ex[j] = expf(e[j] - gmax); ssum += ex[j]; }
    red[t] = ssum; __syncthreads();
    for (int s = 512; s > 0; s >>= 1) {
        if (t < s) red[t] += red[t + s];
        __syncthreads();
    }
    float inv = 1.f / red[0];
#pragma unroll
    for (int j = 0; j < 4; ++j) {
        int idx = t + j * 1024;
        float a = ex[j] * inv;
        alpha[idx] = a;
        asum[idx] += a;
    }
}

// ctx[c] = sum_pos alpha[pos] * feat[c][pos]
__global__ void k_ctx(const float* __restrict__ feat, const float* __restrict__ alpha,
                      float* __restrict__ ctx) {
    __shared__ float red[256];
    int c = blockIdx.x, t = threadIdx.x;
    float s = 0.f;
    for (int p = t; p < NPOS; p += 256) s += alpha[p] * feat[c * NPOS + p];
    red[t] = s; __syncthreads();
    for (int st = 128; st > 0; st >>= 1) {
        if (t < st) red[t] += red[t + st];
        __syncthreads();
    }
    if (t == 0) ctx[c] = red[0];
}

// output GRU + state + classifier + argmax + next embedding (single workgroup)
__global__ void k_out(const float* __restrict__ WihT, const float* __restrict__ WhhT,
                      const float* __restrict__ bih, const float* __restrict__ bhh,
                      const float* __restrict__ WsT, const float* __restrict__ bs,
                      const float* __restrict__ WeT, const float* __restrict__ be,
                      const float* __restrict__ WcT, const float* __restrict__ bc,
                      const float* __restrict__ WvT, const float* __restrict__ bv,
                      const float* __restrict__ emb_table, const float* __restrict__ ctx,
                      const float* __restrict__ h1, float* __restrict__ hout,
                      float* __restrict__ embio, float* __restrict__ out, int step) {
    __shared__ float sCtx[C_DIM], sH1[H_DIM], sH[H_DIM], sE[H_DIM], sState[128];
    __shared__ float sVal[256];
    __shared__ int sIdx[256];
    __shared__ int sWord;
    int t = threadIdx.x;
    for (int i = t; i < C_DIM; i += 256) sCtx[i] = ctx[i];
    sH1[t] = h1[t]; sE[t] = embio[t];
    __syncthreads();
    // GRU(ctx, h1)
    float gr = bih[t], gz = bih[t + 256], gn = bih[t + 512];
    for (int c = 0; c < C_DIM; ++c) {
        float x = sCtx[c];
        const float* wi = WihT + c * 768;
        gr += x * wi[t]; gz += x * wi[t + 256]; gn += x * wi[t + 512];
    }
    float hr = bhh[t], hz = bhh[t + 256], hn = bhh[t + 512];
    for (int c = 0; c < H_DIM; ++c) {
        float hv = sH1[c];
        const float* wh = WhhT + c * 768;
        hr += hv * wh[t]; hz += hv * wh[t + 256]; hn += hv * wh[t + 512];
    }
    float r = sigm(gr + hr), z = sigm(gz + hz);
    float n = fast_tanh(gn + r * hn);
    float hnew = (1.f - z) * n + z * sH1[t];
    sH[t] = hnew; hout[t] = hnew;
    __syncthreads();
    // state (128)
    if (t < 128) {
        float s = bs[t] + be[t] + bc[t];
        for (int i = 0; i < H_DIM; ++i) s += sH[i] * WsT[i * 128 + t];
        for (int i = 0; i < H_DIM; ++i) s += sE[i] * WeT[i * 128 + t];
        for (int i = 0; i < C_DIM; ++i) s += sCtx[i] * WcT[i * 128 + t];
        sState[t] = s;
    }
    __syncthreads();
    // prob (415) + local argmax
    float bmax = -3.0e38f; int bidx = 0;
    for (int v = t; v < V_DIM; v += 256) {
        float p = bv[v];
        for (int i = 0; i < 128; ++i) p += sState[i] * WvT[i * V_DIM + v];
        out[step * V_DIM + v] = p;
        if (p > bmax) { bmax = p; bidx = v; }
    }
    sVal[t] = bmax; sIdx[t] = bidx;
    __syncthreads();
    for (int s = 128; s > 0; s >>= 1) {
        if (t < s) {
            if (sVal[t + s] > sVal[t] ||
                (sVal[t + s] == sVal[t] && sIdx[t + s] < sIdx[t])) {
                sVal[t] = sVal[t + s]; sIdx[t] = sIdx[t + s];
            }
        }
        __syncthreads();
    }
    if (t == 0) sWord = sIdx[0];
    __syncthreads();
    embio[t] = emb_table[sWord * H_DIM + t];
}

extern "C" void kernel_launch(void* const* d_in, const int* in_sizes, int n_in,
                              void* d_out, int out_size, void* d_ws, size_t ws_size,
                              hipStream_t stream) {
    (void)in_sizes; (void)n_in; (void)out_size; (void)ws_size;
    const float* feat    = (const float*)d_in[0];
    const float* imask   = (const float*)d_in[1];
    const float* embT    = (const float*)d_in[2];
    const float* W_init  = (const float*)d_in[3];
    const float* b_init  = (const float*)d_in[4];
    const float* Wih_in  = (const float*)d_in[5];
    const float* Whh_in  = (const float*)d_in[6];
    const float* bih_in  = (const float*)d_in[7];
    const float* bhh_in  = (const float*)d_in[8];
    const float* Wih_out = (const float*)d_in[9];
    const float* Whh_out = (const float*)d_in[10];
    const float* bih_out = (const float*)d_in[11];
    const float* bhh_out = (const float*)d_in[12];
    const float* Wq      = (const float*)d_in[13];
    const float* bq      = (const float*)d_in[14];
    const float* Wproj   = (const float*)d_in[15];
    const float* bproj   = (const float*)d_in[16];
    const float* convk   = (const float*)d_in[17];
    const float* Wcov    = (const float*)d_in[18];
    const float* wa      = (const float*)d_in[19];
    const float* ba      = (const float*)d_in[20];
    const float* Ws      = (const float*)d_in[21];
    const float* bs      = (const float*)d_in[22];
    const float* We      = (const float*)d_in[23];
    const float* be      = (const float*)d_in[24];
    const float* Wc      = (const float*)d_in[25];
    const float* bc      = (const float*)d_in[26];
    const float* Wv      = (const float*)d_in[27];
    const float* bv      = (const float*)d_in[28];
    float* out = (float*)d_out;
    float* w = (float*)d_ws;

    float* cnnt   = w + OFF_CNNT;
    float* keff   = w + OFF_KEFF;
    _Float16* kfr = (_Float16*)(w + OFF_KFRAG);
    float* wihti  = w + OFF_WIHTI;
    float* whhti  = w + OFF_WHHTI;
    float* wihto  = w + OFF_WIHTO;
    float* whhto  = w + OFF_WHHTO;
    float* wqt    = w + OFF_WQT;
    float* wst    = w + OFF_WST;
    float* wet    = w + OFF_WET;
    float* wct    = w + OFF_WCT;
    float* wvt    = w + OFF_WVT;
    float* asum   = w + OFF_ASUM;
    float* alpha  = w + OFF_ALPHA;
    float* energy = w + OFF_ENERGY;
    float* maskv  = w + OFF_MASKV;
    float* avg    = w + OFF_AVG;
    float* h      = w + OFF_H;
    float* h1     = w + OFF_H1;
    float* emb    = w + OFF_EMB;
    float* q      = w + OFF_Q;
    float* ctx    = w + OFF_CTX;

    auto tgrid = [](int R, int Cc) { return dim3((unsigned)((R * Cc + 255) / 256)); };

    // one-time precompute (re-run every call for determinism)
    k_transpose<<<tgrid(768, 256), 256, 0, stream>>>(Wih_in, wihti, 768, 256);
    k_transpose<<<tgrid(768, 256), 256, 0, stream>>>(Whh_in, whhti, 768, 256);
    k_transpose<<<tgrid(768, 684), 256, 0, stream>>>(Wih_out, wihto, 768, 684);
    k_transpose<<<tgrid(768, 256), 256, 0, stream>>>(Whh_out, whhto, 768, 256);
    k_transpose<<<tgrid(512, 256), 256, 0, stream>>>(Wq, wqt, 512, 256);
    k_transpose<<<tgrid(128, 256), 256, 0, stream>>>(Ws, wst, 128, 256);
    k_transpose<<<tgrid(128, 256), 256, 0, stream>>>(We, wet, 128, 256);
    k_transpose<<<tgrid(128, 684), 256, 0, stream>>>(Wc, wct, 128, 684);
    k_transpose<<<tgrid(415, 128), 256, 0, stream>>>(Wv, wvt, 415, 128);

    k_mask<<<16, 256, 0, stream>>>(imask, maskv);
    k_avg<<<C_DIM, 256, 0, stream>>>(feat, maskv, avg);
    k_h0<<<1, 256, 0, stream>>>(avg, W_init, b_init, h);
    k_init<<<16, 256, 0, stream>>>(embT, asum, emb);
    k_keff<<<(KPAD * A_DIM) / 256, 256, 0, stream>>>(convk, Wcov, keff);
    k_kfrag<<<(32 * 4 * 32 * 16) / 256, 256, 0, stream>>>(keff, kfr);
    k_cnntrans<<<dim3(16, 32), 256, 0, stream>>>(feat, Wproj, bproj, cnnt);

    for (int t = 0; t < TSTEPS; ++t) {
        k_gru_in<<<1, 256, 0, stream>>>(wihti, whhti, bih_in, bhh_in, wqt, bq,
                                        emb, h, h1, q);
        k_attend<<<NPOS / 16, 256, 0, stream>>>(kfr, asum, q, cnnt, wa, ba,
                                                maskv, energy);
        k_softmax<<<1, 1024, 0, stream>>>(energy, alpha, asum);
        k_ctx<<<C_DIM, 256, 0, stream>>>(feat, alpha, ctx);
        k_out<<<1, 256, 0, stream>>>(wihto, whhto, bih_out, bhh_out,
                                     wst, bs, wet, be, wct, bc, wvt, bv,
                                     embT, ctx, h1, h, emb, out, t);
    }
}
